// LlamaDecoderLayer_1786706395700
// MI455X (gfx1250) — matrix operand order repaired and checked by reading, not compiled, not run on hardware
//
#include <hip/hip_runtime.h>
#include <hip/hip_bf16.h>
#include <cstdint>
#include <cstddef>

// ---------------- problem constants (match reference) ----------------
constexpr int B_  = 4;
constexpr int L_  = 1024;
constexpr int D_  = 2048;
constexpr int H_  = 16;
constexpr int I_  = 8192;
constexpr int DH_ = D_ / H_;     // 128
constexpr int M_  = B_ * L_;     // 4096 token rows
constexpr float EPS_ = 1e-6f;

typedef _Float16 half_t;
typedef __attribute__((ext_vector_type(16))) _Float16 v16h;
typedef __attribute__((ext_vector_type(8)))  _Float16 v8h;
typedef __attribute__((ext_vector_type(8)))  float    v8f;

#define BKT 32
#define LDSPAD 8   // keeps 16B alignment, spreads LDS banks

// CDNA5 async global->LDS path (ASYNCcnt), with compile-safe fallback.
#if defined(__gfx1250__) && \
    __has_builtin(__builtin_amdgcn_global_load_async_to_lds_b128) && \
    __has_builtin(__builtin_amdgcn_s_wait_asynccnt)
#define USE_ASYNC_LDS 1
// builtin expects (int4 AS1* src, int4 AS3* lds, imm offset, imm cpol)
typedef int i128_t __attribute__((vector_size(16)));
typedef i128_t __attribute__((address_space(1)))* gvec_p;
typedef i128_t __attribute__((address_space(3)))* lvec_p;
#else
#define USE_ASYNC_LDS 0
#endif

// =====================================================================
// f16 WMMA GEMM:  C[z] = alpha * A[z] (MxK) @ B[z] (NxK)^T  (+ addend)
// Template tile: TBM x TBN, 8 wave32 arranged 2(M) x 4(N),
// wave tile (TBM/2) x (TBN/4):  <128,256> -> 64x64 = 16 WMMA accumulators.
// Double-buffered LDS; async global->LDS staging when available.
//   C += (z/czdiv)*strideC1 + (z%czdiv)*strideC2
//   addend (optional, f32): C += addend[(z/addZdiv)*addStride + m*addLd + n]
// =====================================================================
template<int TBM, int TBN>
__global__ __launch_bounds__(256)
void gemm_f16_nt_kernel(const half_t* __restrict__ A,
                        const half_t* __restrict__ Bm,
                        float* __restrict__ C,
                        int K, int lda, int ldb, int ldc,
                        long long strideA, long long strideB,
                        int czdiv, long long strideC1, long long strideC2,
                        float alpha,
                        const float* __restrict__ addend,
                        long long addStride, int addZdiv, int addLd)
{
    constexpr int WROWS = TBM / 2;     // rows per wave
    constexpr int WCOLS = TBN / 4;     // cols per wave
    constexpr int MT = WROWS / 16;     // 16x16 tiles (M) per wave
    constexpr int NT = WCOLS / 16;     // 16x16 tiles (N) per wave

    __shared__ __attribute__((aligned(32))) half_t Alds[2][TBM][BKT + LDSPAD];
    __shared__ __attribute__((aligned(32))) half_t Blds[2][TBN][BKT + LDSPAD];

    const int z = blockIdx.z;
    A  += (long long)z * strideA;
    Bm += (long long)z * strideB;
    C  += (long long)(z / czdiv) * strideC1 + (long long)(z % czdiv) * strideC2;
    const float* addp = addend ? addend + (long long)(z / addZdiv) * addStride
                               : (const float*)nullptr;

    const int tid   = threadIdx.x;
    const int lane  = tid & 31;
    const int wave  = tid >> 5;
    const int waveM = wave & 1;        // 0..1
    const int waveN = wave >> 1;       // 0..3
    const long long tileM0 = (long long)blockIdx.x * TBM;
    const long long tileN0 = (long long)blockIdx.y * TBN;

    v8f acc[MT][NT] = {};

    // per-lane fragment geometry (CDNA5 ISA 7.12.2, wave32)
    const int row16 = lane & 15;
    const int aKlo  = (lane < 16) ? 0 : 8;    // A: halves {klo..klo+7, 16+klo..}
    const int bKs   = (lane < 16) ? 0 : 8;    // B: halves {bKs..bKs+7, 16+bKs..}

    // stage one K-slab (TBM x 32 of A, TBN x 32 of B) into LDS buffer `buf`
    auto stage = [&](int buf, int k0) {
        #pragma unroll
        for (int c = tid; c < TBM * 4; c += 256) {         // A: 16B chunks
            const int r  = c >> 2;
            const int ch = (c & 3) * 8;
            const half_t* g = &A[(tileM0 + r) * (long long)lda + k0 + ch];
            half_t* l = &Alds[buf][r][ch];
#if USE_ASYNC_LDS
            __builtin_amdgcn_global_load_async_to_lds_b128(
                (gvec_p)g, (lvec_p)l, 0, 0);
#else
            *(uint4*)l = *(const uint4*)g;
#endif
        }
        #pragma unroll
        for (int c = tid; c < TBN * 4; c += 256) {         // B: 16B chunks
            const int r  = c >> 2;
            const int ch = (c & 3) * 8;
            const half_t* g = &Bm[(tileN0 + r) * (long long)ldb + k0 + ch];
            half_t* l = &Blds[buf][r][ch];
#if USE_ASYNC_LDS
            __builtin_amdgcn_global_load_async_to_lds_b128(
                (gvec_p)g, (lvec_p)l, 0, 0);
#else
            *(uint4*)l = *(const uint4*)g;
#endif
        }
    };

    stage(0, 0);
    int cur = 0;

    for (int k0 = 0; k0 < K; k0 += BKT) {
#if USE_ASYNC_LDS
        __builtin_amdgcn_s_wait_asynccnt(0);   // staging of `cur` landed
#endif
        __syncthreads();

        if (k0 + BKT < K) stage(cur ^ 1, k0 + BKT);   // prefetch next slab

        // ---- B fragments for this wave's NT column tiles ----
        v16h bfrag[NT];
        #pragma unroll
        for (int nt = 0; nt < NT; ++nt) {
            const int n = waveN * WCOLS + nt * 16 + row16;
            v8h blo = *(const v8h*)&Blds[cur][n][bKs];
            v8h bhi = *(const v8h*)&Blds[cur][n][16 + bKs];
            bfrag[nt] = __builtin_shufflevector(blo, bhi,
                        0,1,2,3,4,5,6,7,8,9,10,11,12,13,14,15);
        }

        #pragma unroll
        for (int mt = 0; mt < MT; ++mt) {
            const int r = waveM * WROWS + mt * 16 + row16;
            v8h alo = *(const v8h*)&Alds[cur][r][aKlo];
            v8h ahi = *(const v8h*)&Alds[cur][r][16 + aKlo];
            v16h afrag = __builtin_shufflevector(alo, ahi,
                         0,1,2,3,4,5,6,7,8,9,10,11,12,13,14,15);
            #pragma unroll
            for (int nt = 0; nt < NT; ++nt) {
                acc[mt][nt] = __builtin_amdgcn_wmma_f32_16x16x32_f16(
                    false, afrag, false, bfrag[nt],
                    (short)0, acc[mt][nt], false, false);
            }
        }

        __syncthreads();   // all waves done reading `cur` before re-stage
        cur ^= 1;
    }

    // ---- epilogue: C/D layout = VGPR r holds (M = r or 8+r, N = lane%16) ----
    const int cn   = row16;
    const int cmhi = (lane < 16) ? 0 : 8;
    #pragma unroll
    for (int mt = 0; mt < MT; ++mt) {
        #pragma unroll
        for (int nt = 0; nt < NT; ++nt) {
            const long long gm0 = tileM0 + waveM * WROWS + mt * 16 + cmhi;
            const long long gn  = tileN0 + waveN * WCOLS + nt * 16 + cn;
            #pragma unroll
            for (int r = 0; r < 8; ++r) {
                float val = alpha * acc[mt][nt][r];
                if (addp) val += addp[(gm0 + r) * (long long)addLd + gn];
                C[(gm0 + r) * (long long)ldc + gn] = val;
            }
        }
    }
}

// =====================================================================
// Support kernels
// =====================================================================
__global__ void cvt_f32_f16_kernel(const float* __restrict__ s,
                                   half_t* __restrict__ d, size_t n)
{
    size_t i = (size_t)blockIdx.x * blockDim.x + threadIdx.x;
    size_t st = (size_t)gridDim.x * blockDim.x;
    for (; i < n; i += st) d[i] = (half_t)s[i];
}

__global__ __launch_bounds__(256)
void rmsnorm_f16_kernel(const float* __restrict__ x,
                        const float* __restrict__ w,
                        half_t* __restrict__ out)
{
    const size_t row = blockIdx.x;
    const float* xr = x + row * (size_t)D_;
    float ss = 0.f;
    for (int i = threadIdx.x; i < D_; i += 256) { float v = xr[i]; ss += v * v; }
    __shared__ float red[256];
    red[threadIdx.x] = ss; __syncthreads();
    for (int s = 128; s > 0; s >>= 1) {
        if (threadIdx.x < s) red[threadIdx.x] += red[threadIdx.x + s];
        __syncthreads();
    }
    const float inv = rsqrtf(red[0] / (float)D_ + EPS_);
    half_t* orow = out + row * (size_t)D_;
    for (int i = threadIdx.x; i < D_; i += 256)
        orow[i] = (half_t)(xr[i] * inv * w[i]);
}

// RoPE + repack [B,L,H,DH] f32 -> [B,H,L,DH] f16
__global__ void rope_repack_kernel(const float* __restrict__ src,
                                   const float* __restrict__ cs,
                                   const float* __restrict__ sn,
                                   half_t* __restrict__ dst)
{
    size_t t = (size_t)blockIdx.x * blockDim.x + threadIdx.x;
    if (t >= (size_t)M_ * D_) return;
    const int d = (int)(t % DH_); size_t r = t / DH_;
    const int h = (int)(r % H_);  r /= H_;
    const int l = (int)(r % L_);
    const int b = (int)(r / L_);
    const size_t srow = (((size_t)b * L_ + l) * H_ + h) * DH_;
    const float v   = src[srow + d];
    const float rot = (d < DH_ / 2) ? -src[srow + d + DH_ / 2]
                                    :  src[srow + d - DH_ / 2];
    const size_t ci = ((size_t)b * L_ + l) * DH_ + d;
    const float o = v * cs[ci] + rot * sn[ci];
    dst[(((size_t)b * H_ + h) * L_ + l) * DH_ + d] = (half_t)o;
}

// V repack: [B,L,H,DH] f32 -> V^T [B,H,DH,L] f16
__global__ void vtrans_kernel(const float* __restrict__ src,
                              half_t* __restrict__ dst)
{
    size_t t = (size_t)blockIdx.x * blockDim.x + threadIdx.x;
    if (t >= (size_t)M_ * D_) return;
    const int l = (int)(t % L_);  size_t r = t / L_;
    const int d = (int)(r % DH_); r /= DH_;
    const int h = (int)(r % H_);
    const int b = (int)(r / H_);
    dst[t] = (half_t)src[(((size_t)b * L_ + l) * H_ + h) * DH_ + d];
}

// row softmax over L in-place (f32 probs stay in d_out) + f16 copy for P·V
__global__ __launch_bounds__(256)
void softmax_kernel(float* __restrict__ logits, half_t* __restrict__ a16)
{
    const size_t row = blockIdx.x;
    float* p  = logits + row * (size_t)L_;
    half_t* q = a16    + row * (size_t)L_;
    __shared__ float red[256];
    float mx = -3.4e38f;
    for (int i = threadIdx.x; i < L_; i += 256) mx = fmaxf(mx, p[i]);
    red[threadIdx.x] = mx; __syncthreads();
    for (int s = 128; s > 0; s >>= 1) {
        if (threadIdx.x < s) red[threadIdx.x] = fmaxf(red[threadIdx.x], red[threadIdx.x + s]);
        __syncthreads();
    }
    mx = red[0]; __syncthreads();
    float sum = 0.f;
    for (int i = threadIdx.x; i < L_; i += 256) {
        float e = __expf(p[i] - mx); p[i] = e; sum += e;
    }
    red[threadIdx.x] = sum; __syncthreads();
    for (int s = 128; s > 0; s >>= 1) {
        if (threadIdx.x < s) red[threadIdx.x] += red[threadIdx.x + s];
        __syncthreads();
    }
    const float inv = 1.f / red[0];
    for (int i = threadIdx.x; i < L_; i += 256) {
        float v = p[i] * inv; p[i] = v; q[i] = (half_t)v;
    }
}

__global__ void silu_mul_kernel(const float* __restrict__ g,
                                const float* __restrict__ u,
                                half_t* __restrict__ out, size_t n)
{
    size_t t = (size_t)blockIdx.x * blockDim.x + threadIdx.x;
    if (t >= n) return;
    const float gv = g[t];
    out[t] = (half_t)(gv / (1.f + __expf(-gv)) * u[t]);
}

// =====================================================================
// Host side
// =====================================================================
template<int TBM, int TBN>
static void launch_gemm(hipStream_t stream,
                        const half_t* A, const half_t* Bm, float* C,
                        int Mi, int Ni, int Ki, int lda, int ldb, int ldc, int nz,
                        long long sA, long long sB,
                        int czdiv, long long sC1, long long sC2,
                        float alpha, const float* addp,
                        long long addS, int addZdiv, int addLd)
{
    dim3 g((unsigned)(Mi / TBM), (unsigned)(Ni / TBN), (unsigned)nz);
    gemm_f16_nt_kernel<TBM, TBN><<<g, dim3(256), 0, stream>>>(
        A, Bm, C, Ki, lda, ldb, ldc, sA, sB, czdiv, sC1, sC2,
        alpha, addp, addS, addZdiv, addLd);
}

extern "C" void kernel_launch(void* const* d_in, const int* in_sizes, int n_in,
                              void* d_out, int out_size, void* d_ws, size_t ws_size,
                              hipStream_t stream)
{
    (void)in_sizes; (void)n_in; (void)out_size; (void)ws_size;

    const float* hidden = (const float*)d_in[0];   // [B,L,D]
    const float* mask   = (const float*)d_in[1];   // [B,1,L,L]
    const float* cosb   = (const float*)d_in[2];   // [B,L,DH]
    const float* sinb   = (const float*)d_in[3];   // [B,L,DH]
    const float* q_w    = (const float*)d_in[4];   // [D,D]
    const float* k_w    = (const float*)d_in[5];
    const float* v_w    = (const float*)d_in[6];
    const float* o_w    = (const float*)d_in[7];
    const float* gate_w = (const float*)d_in[8];   // [I,D]
    const float* up_w   = (const float*)d_in[9];   // [I,D]
    const float* down_w = (const float*)d_in[10];  // [D,I]
    const float* ln1_w  = (const float*)d_in[11];
    const float* ln2_w  = (const float*)d_in[12];

    float* out_hid  = (float*)d_out;                       // [B,L,D]
    float* attn_out = (float*)d_out + (size_t)M_ * D_;     // [B,H,L,L]

    char* base = (char*)d_ws;
    size_t off = 0;
    auto alloc = [&](size_t bytes) -> char* {
        char* p = base + off;
        off += (bytes + 255) & ~(size_t)255;
        return p;
    };
    const size_t DD = (size_t)D_ * D_, ID = (size_t)I_ * D_;
    const size_t MD = (size_t)M_ * D_, MI = (size_t)M_ * I_;
    const size_t HLL = (size_t)B_ * H_ * L_ * L_;

    half_t* wq16  = (half_t*)alloc(DD * 2);
    half_t* wk16  = (half_t*)alloc(DD * 2);
    half_t* wv16  = (half_t*)alloc(DD * 2);
    half_t* wo16  = (half_t*)alloc(DD * 2);
    half_t* wg16  = (half_t*)alloc(ID * 2);
    half_t* wu16  = (half_t*)alloc(ID * 2);
    half_t* wd16  = (half_t*)alloc(ID * 2);
    half_t* hn16  = (half_t*)alloc(MD * 2);
    float*  qf32  = (float*) alloc(MD * 4);
    float*  kf32  = (float*) alloc(MD * 4);
    float*  vf32  = (float*) alloc(MD * 4);
    half_t* q16   = (half_t*)alloc(MD * 2);     // [B,H,L,DH]
    half_t* k16   = (half_t*)alloc(MD * 2);     // [B,H,L,DH]
    half_t* vT16  = (half_t*)alloc(MD * 2);     // [B,H,DH,L]
    half_t* p16   = (half_t*)alloc(HLL * 2);    // f16 attn probs
    float*  ctx32 = (float*) alloc(MD * 4);     // [B,L,D]
    half_t* ctx16 = (half_t*)alloc(MD * 2);
    float*  h1    = (float*) alloc(MD * 4);
    half_t* h2n16 = (half_t*)alloc(MD * 2);
    float*  g32   = (float*) alloc(MI * 4);
    float*  u32   = (float*) alloc(MI * 4);
    half_t* act16 = (half_t*)alloc(MI * 2);

    const dim3 blk(256);
    auto nblk = [](size_t n) { return dim3((unsigned)((n + 255) / 256)); };

    // 1) weights -> f16
    cvt_f32_f16_kernel<<<nblk(DD), blk, 0, stream>>>(q_w,    wq16, DD);
    cvt_f32_f16_kernel<<<nblk(DD), blk, 0, stream>>>(k_w,    wk16, DD);
    cvt_f32_f16_kernel<<<nblk(DD), blk, 0, stream>>>(v_w,    wv16, DD);
    cvt_f32_f16_kernel<<<nblk(DD), blk, 0, stream>>>(o_w,    wo16, DD);
    cvt_f32_f16_kernel<<<nblk(ID), blk, 0, stream>>>(gate_w, wg16, ID);
    cvt_f32_f16_kernel<<<nblk(ID), blk, 0, stream>>>(up_w,   wu16, ID);
    cvt_f32_f16_kernel<<<nblk(ID), blk, 0, stream>>>(down_w, wd16, ID);

    // 2) rmsnorm1
    rmsnorm_f16_kernel<<<dim3(M_), blk, 0, stream>>>(hidden, ln1_w, hn16);

    // 3) Q/K/V projections (4096 x 2048 x 2048)
    launch_gemm<128,256>(stream, hn16, wq16, qf32, M_, D_, D_, D_, D_, D_, 1,
                         0, 0, 1, 0, 0, 1.f, nullptr, 0, 1, D_);
    launch_gemm<128,256>(stream, hn16, wk16, kf32, M_, D_, D_, D_, D_, D_, 1,
                         0, 0, 1, 0, 0, 1.f, nullptr, 0, 1, D_);
    launch_gemm<128,256>(stream, hn16, wv16, vf32, M_, D_, D_, D_, D_, D_, 1,
                         0, 0, 1, 0, 0, 1.f, nullptr, 0, 1, D_);

    // 4) RoPE + head-major repack; V transpose
    rope_repack_kernel<<<nblk(MD), blk, 0, stream>>>(qf32, cosb, sinb, q16);
    rope_repack_kernel<<<nblk(MD), blk, 0, stream>>>(kf32, cosb, sinb, k16);
    vtrans_kernel<<<nblk(MD), blk, 0, stream>>>(vf32, vT16);

    // 5) scores = Q K^T / sqrt(DH) + mask  (z = b*H+h, 1024x1024x128 each)
    const float scale = 0.08838834764831845f;  // 1/sqrt(128)
    launch_gemm<128,256>(stream, q16, k16, attn_out, L_, L_, DH_, DH_, DH_, L_,
                         B_ * H_, (long long)L_ * DH_, (long long)L_ * DH_,
                         1, (long long)L_ * L_, 0,
                         scale, mask, (long long)L_ * L_, H_, L_);

    // 6) softmax rows (in-place f32 + f16 copy)
    softmax_kernel<<<dim3((unsigned)(B_ * H_ * L_)), blk, 0, stream>>>(attn_out, p16);

    // 7) ctx = P V  (N = DH = 128 -> 128x128 tile variant)
    launch_gemm<128,128>(stream, p16, vT16, ctx32, L_, DH_, L_, L_, L_, D_,
                         B_ * H_, (long long)L_ * L_, (long long)DH_ * L_,
                         H_, (long long)L_ * D_, (long long)DH_,
                         1.f, nullptr, 0, 1, D_);
    cvt_f32_f16_kernel<<<nblk(MD), blk, 0, stream>>>(ctx32, ctx16, MD);

    // 8) h1 = hidden + ctx @ o_w^T (residual fused)
    launch_gemm<128,256>(stream, ctx16, wo16, h1, M_, D_, D_, D_, D_, D_, 1,
                         0, 0, 1, 0, 0, 1.f, hidden, 0, 1, D_);

    // 9) rmsnorm2
    rmsnorm_f16_kernel<<<dim3(M_), blk, 0, stream>>>(h1, ln2_w, h2n16);

    // 10) gate / up (4096 x 8192 x 2048)
    launch_gemm<128,256>(stream, h2n16, wg16, g32, M_, I_, D_, D_, D_, I_, 1,
                         0, 0, 1, 0, 0, 1.f, nullptr, 0, 1, D_);
    launch_gemm<128,256>(stream, h2n16, wu16, u32, M_, I_, D_, D_, D_, I_, 1,
                         0, 0, 1, 0, 0, 1.f, nullptr, 0, 1, D_);

    // 11) act = silu(gate) * up
    silu_mul_kernel<<<nblk(MI), blk, 0, stream>>>(g32, u32, act16, MI);

    // 12) out = h1 + act @ down_w^T (4096 x 2048 x 8192, residual fused)
    launch_gemm<128,256>(stream, act16, wd16, out_hid, M_, D_, I_, I_, I_, D_, 1,
                         0, 0, 1, 0, 0, 1.f, h1, 0, 1, D_);
}
